// Cabasc_64905545777226
// MI455X (gfx1250) — matrix operand order repaired and checked by reading, not compiled, not run on hardware
//
#include <hip/hip_runtime.h>

#define B_  128
#define L_  512
#define LA_ 5
#define D_  300
#define H_  300
#define P_  3
#define DK  320      // K padded to 10 x 32 for f16 WMMA
#define NP  912      // GRU packed N: 19 hidden tiles * 3 gates * 16
#define NK  304      // Wk packed N: 19 tiles * 16
#define KXS 304      // kx row stride (floats)

typedef __attribute__((ext_vector_type(16))) _Float16 v16h;
typedef __attribute__((ext_vector_type(8)))  _Float16 v8h;
typedef __attribute__((ext_vector_type(8)))  float    v8f;

__device__ __forceinline__ float sigmoidf(float x){ return 1.f/(1.f+__expf(-x)); }

// A-matrix 16x32 f16 fragment from row-major (lda halfs) storage.
// ISA layout: lanes 0-15 row=lane, v0..3 = K 0..7, v4..7 = K 16..23;
// lanes 16-31 same rows, v0..3 = K 8..15, v4..7 = K 24..31.
__device__ __forceinline__ v16h a_frag(const _Float16* A, int lda, int kt, int lane){
  int m    = lane & 15;
  int koff = (lane >> 4) << 3;          // 0 or 8
  const _Float16* r = A + m*lda + kt*32;
  v8h lo = *(const v8h*)(r + koff);
  v8h hi = *(const v8h*)(r + 16 + koff);
  return __builtin_shufflevector(lo, hi, 0,1,2,3,4,5,6,7,8,9,10,11,12,13,14,15);
}
// B-matrix 32x16 f16 fragment; weights pre-packed row-major W[n][k] (= Wᵀ as B).
// ISA layout: lane n holds column n; lanes 0-15 K 0..7,16..23; lanes 16-31 K 8..15,24..31 (same as A).
__device__ __forceinline__ v16h b_frag(const _Float16* W, int ldw, int n0, int lane, int kt){
  const _Float16* r = W + (size_t)(n0 + (lane & 15))*ldw + kt*32 + ((lane >> 4) << 3);
  v8h lo = *(const v8h*)r;
  v8h hi = *(const v8h*)(r + 16);
  return __builtin_shufflevector(lo, hi, 0,1,2,3,4,5,6,7,8,9,10,11,12,13,14,15);
}
__device__ __forceinline__ v8f wmma_f16(v16h a, v16h b, v8f c){
  return __builtin_amdgcn_wmma_f32_16x16x32_f16(false, a, false, b, (short)0, c, false, false);
}

// ---------------- lengths ----------------
__global__ void lens_kernel(const int* text, const int* aidx, const int* xl, const int* xr,
                            int* lens){
  int tid = threadIdx.x; if (tid >= 512) return;
  int b = tid >> 2, which = tid & 3;
  const int* p; int n;
  if (which == 0){ p = text + b*L_;  n = L_;  }
  else if (which == 1){ p = aidx + b*LA_; n = LA_; }
  else if (which == 2){ p = xl + b*L_;   n = L_;  }
  else { p = xr + b*L_; n = L_; }
  int c = 0;
  for (int i = 0; i < n; ++i) c += (p[i] != 0);
  lens[which*B_ + b] = c;
}

// ---------------- weight packing ----------------
// GRU pack: packed row p = ht*48 + gate*16 + u  ->  orig row gate*300 + ht*16 + u
__global__ void pack_gru_kernel(const float* WihL, const float* WhhL,
                                const float* WihR, const float* WhhR, _Float16* out){
  size_t i = (size_t)blockIdx.x*blockDim.x + threadIdx.x;
  if (i >= (size_t)4*NP*DK) return;
  int mat = i / (NP*DK);
  int rem = i - (size_t)mat*NP*DK;
  int p = rem / DK, k = rem % DK;
  int ht = p / 48, g = (p % 48) / 16, u = p & 15;
  int hr = ht*16 + u;
  float v = 0.f;
  if (hr < H_ && k < D_){
    const float* W = (mat==0)?WihL:(mat==1)?WhhL:(mat==2)?WihR:WhhR;
    v = W[(size_t)(g*H_ + hr)*D_ + k];
  }
  out[i] = (_Float16)v;
}
__global__ void pack_bias_kernel(const float* biL, const float* bhL,
                                 const float* biR, const float* bhR, float* out){
  int i = blockIdx.x*blockDim.x + threadIdx.x;
  if (i >= 4*NP) return;
  int arr = i / NP, p = i % NP;
  int ht = p / 48, g = (p % 48) / 16, u = p & 15;
  int hr = ht*16 + u;
  float v = 0.f;
  if (hr < H_){
    const float* bsrc = (arr==0)?biL:(arr==1)?bhL:(arr==2)?biR:bhR;
    v = bsrc[g*H_ + hr];
  }
  out[i] = v;
}
__global__ void pack_wk_kernel(const float* Wk, _Float16* out){
  int i = blockIdx.x*blockDim.x + threadIdx.x;
  if (i >= NK*DK) return;
  int n = i / DK, k = i % DK;
  out[i] = (_Float16)((n < D_ && k < D_) ? Wk[(size_t)n*D_ + k] : 0.f);
}

// ---------------- aspect mean + qdot ----------------
__global__ void aspect_kernel(const int* aidx, const float* emb, const int* lens,
                              const float* Wq, const float* bq, const float* w_att,
                              float* qdot){
  __shared__ float asp[304];
  __shared__ float red[512];
  int b = blockIdx.x, d = threadIdx.x;
  if (d < D_){
    float s = 0.f;
    for (int i = 0; i < LA_; ++i) s += emb[(size_t)aidx[b*LA_ + i]*D_ + d];
    asp[d] = s / (float)lens[B_ + b];
  } else if (d < 304) asp[d] = 0.f;
  __syncthreads();
  float contrib = 0.f;
  if (d < D_){
    float q = bq[d];
    for (int k = 0; k < D_; ++k) q += asp[k]*Wq[(size_t)d*D_ + k];
    contrib = q * w_att[D_ + d];
  }
  red[d] = contrib; __syncthreads();
  for (int o = 256; o; o >>= 1){ if (d < o) red[d] += red[d + o]; __syncthreads(); }
  if (d == 0) qdot[b] = red[0];
}

// ---------------- persistent GRU (WMMA) ----------------
__global__ __launch_bounds__(512) void gru_kernel(
    const int* xl, const int* xr, const float* emb,
    const _Float16* packW, const float* pbias, const int* lens,
    const float* wl, const float* blp, const float* wr, const float* brp,
    float* attn){
  __shared__ _Float16 ax[16][DK];        // x_t rows (f16, zero-padded K)
  __shared__ _Float16 ah[2][16][DK];     // h rows, double buffered
  __shared__ float    hf[16][H_];        // h in f32
  __shared__ float    biasI[NP], biasH[NP];
  __shared__ float    wvec[304];
  __shared__ int      idxb[16];
  __shared__ int      lenb[16];
  __shared__ float    sdot[16];

  int tid = threadIdx.x, lane = tid & 31, wave = tid >> 5;
  int dir = blockIdx.x >> 3;
  int bbase = (blockIdx.x & 7) * 16;
  const int*      xind = dir ? xr : xl;
  const _Float16* Wih  = packW + (size_t)(dir*2 + 0)*NP*DK;
  const _Float16* Whh  = packW + (size_t)(dir*2 + 1)*NP*DK;
  const float*    pbih = pbias + (dir*2 + 0)*NP;
  const float*    pbhh = pbias + (dir*2 + 1)*NP;
  const int*      len  = lens + (dir ? 3*B_ : 2*B_);
  const float*    wv   = dir ? wr : wl;
  float battn = dir ? brp[0] : blp[0];

  for (int i = tid; i < 16*DK; i += 512){
    int m = i / DK, k = i % DK;
    ax[m][k] = (_Float16)0.f; ah[0][m][k] = (_Float16)0.f; ah[1][m][k] = (_Float16)0.f;
  }
  for (int i = tid; i < 16*H_; i += 512) hf[i / H_][i % H_] = 0.f;
  for (int i = tid; i < NP; i += 512){ biasI[i] = pbih[i]; biasH[i] = pbhh[i]; }
  for (int i = tid; i < 304; i += 512) wvec[i] = (i < D_) ? wv[i] : 0.f;
  if (tid < 16) lenb[tid] = len[bbase + tid];
  __syncthreads();

  const v8f vzero = {0.f,0.f,0.f,0.f,0.f,0.f,0.f,0.f};
  int cur = 0;

  for (int t = 0; t < L_; ++t){
    if (tid < 16) idxb[tid] = xind[(bbase + tid)*L_ + t];
    else if (tid < 32) sdot[tid - 16] = 0.f;
    __syncthreads();
    // gather x_t rows (embedding in L2) -> f16 LDS
    for (int i = tid; i < 16*D_; i += 512){
      int m = i / D_, k = i - m*D_;
      ax[m][k] = (_Float16)emb[(size_t)idxb[m]*D_ + k];
    }
    __syncthreads();

    for (int ht = wave; ht < 19; ht += 16){
      int nb = ht*48;                 // packed rows: [r|z|n] x16 for this hidden tile
      v8f ri = vzero, rh = vzero, zi = vzero, zh = vzero, ni = vzero, nh = vzero;
      for (int kt = 0; kt < 10; ++kt){
        v16h afx = a_frag(&ax[0][0],        DK, kt, lane);
        v16h afh = a_frag(&ah[cur][0][0],   DK, kt, lane);
        __builtin_prefetch(Wih + (size_t)nb*DK + (kt + 1)*32, 0, 1);
        __builtin_prefetch(Whh + (size_t)nb*DK + (kt + 1)*32, 0, 1);
        v16h bri = b_frag(Wih, DK, nb +  0, lane, kt);
        v16h brh = b_frag(Whh, DK, nb +  0, lane, kt);
        ri = wmma_f16(afx, bri, ri);  rh = wmma_f16(afh, brh, rh);
        v16h bzi = b_frag(Wih, DK, nb + 16, lane, kt);
        v16h bzh = b_frag(Whh, DK, nb + 16, lane, kt);
        zi = wmma_f16(afx, bzi, zi);  zh = wmma_f16(afh, bzh, zh);
        v16h bni = b_frag(Wih, DK, nb + 32, lane, kt);
        v16h bnh = b_frag(Whh, DK, nb + 32, lane, kt);
        ni = wmma_f16(afx, bni, ni);  nh = wmma_f16(afh, bnh, nh);
      }
      // gate math in registers; D layout: col = lane&15, rows = (lane>=16?8:0)+r
      int jc = lane & 15;
      int j  = ht*16 + jc;
      int rb = (lane >> 4) << 3;
      float bir = biasI[nb + jc],      bhr = biasH[nb + jc];
      float biz = biasI[nb + 16 + jc], bhz = biasH[nb + 16 + jc];
      float bin = biasI[nb + 32 + jc], bhn = biasH[nb + 32 + jc];
      float wlj = (j < H_) ? wvec[j] : 0.f;
      #pragma unroll
      for (int r2 = 0; r2 < 8; ++r2){
        int m = rb + r2;
        bool act = (t < lenb[m]);
        float rr = sigmoidf(ri[r2] + rh[r2] + bir + bhr);
        float zz = sigmoidf(zi[r2] + zh[r2] + biz + bhz);
        float nn = tanhf(ni[r2] + bin + rr*(nh[r2] + bhn));
        float hp = (j < H_) ? hf[m][j] : 0.f;
        float hn2 = (1.f - zz)*nn + zz*hp;
        float hout = act ? hn2 : hp;
        float oo   = (act && j < H_) ? hn2 : 0.f;
        if (j < H_){ hf[m][j] = hout; ah[cur ^ 1][m][j] = (_Float16)hout; }
        float c = oo * wlj;                     // attn dot contribution
        for (int s = 8; s; s >>= 1) c += __shfl_xor(c, s, 16);
        if ((lane & 15) == 0) atomicAdd(&sdot[m], c);
      }
    }
    __syncthreads();
    if (tid < 16)
      attn[((size_t)dir*B_ + bbase + tid)*L_ + t] = sigmoidf(sdot[tid] + battn) + 0.5f;
    cur ^= 1;
    __syncthreads();
  }
}

// ---------------- positional weights ----------------
__global__ void wpos_kernel(const int* lens, const float* attn, float* w){
  int b = blockIdx.x, l = threadIdx.x;
  int ml = lens[b], al = lens[B_ + b], ll = lens[2*B_ + b];
  int as = ll - al, ae = ll;
  int ri = l - as; ri = ri < 0 ? 0 : (ri > L_ - 1 ? L_ - 1 : ri);
  float ar  = attn[(size_t)(B_ + b)*L_ + ri];
  float alv = attn[(size_t)b*L_ + l];
  float wv = (l < as) ? alv : ((l < ae) ? alv + ar : ar);
  if (l >= ml) wv = 1.f;
  w[(size_t)b*L_ + l] = wv;
}

// ---------------- v_s = mean(w * memory) ----------------
__global__ void vs_kernel(const int* text, const float* emb, const float* w,
                          const int* lens, float* vs){
  int b = blockIdx.x, d = threadIdx.x;
  if (d >= D_) return;
  float s = 0.f;
  for (int l = 0; l < L_; ++l)
    s += w[(size_t)b*L_ + l] * emb[(size_t)text[b*L_ + l]*D_ + d];
  vs[(size_t)b*NK + d] = s / (float)lens[b];
}

// ---------------- kx = w * (memory @ Wkᵀ) + bk  (WMMA) ----------------
__global__ __launch_bounds__(128) void kx_kernel(const int* text, const float* emb,
                                                 const _Float16* Wkp, const float* bk,
                                                 const float* w, float* kx){
  __shared__ _Float16 axs[16][DK];
  __shared__ int idxb[16];
  int tid = threadIdx.x, lane = tid & 31, wave = tid >> 5;
  int row0 = blockIdx.x * 16;                      // global row in [0, B*L)
  if (tid < 16) idxb[tid] = text[row0 + tid];
  __syncthreads();
  for (int i = tid; i < 16*DK; i += 128){
    int m = i / DK, k = i - m*DK;
    axs[m][k] = (_Float16)((k < D_) ? emb[(size_t)idxb[m]*D_ + k] : 0.f);
  }
  __syncthreads();
  const v8f vzero = {0.f,0.f,0.f,0.f,0.f,0.f,0.f,0.f};
  for (int nt = wave; nt < 19; nt += 4){
    v8f acc = vzero;
    for (int kt = 0; kt < 10; ++kt){
      __builtin_prefetch(Wkp + (size_t)(nt*16)*DK + (kt + 1)*32, 0, 1);
      v16h af = a_frag(&axs[0][0], DK, kt, lane);
      v16h bf = b_frag(Wkp, DK, nt*16, lane, kt);
      acc = wmma_f16(af, bf, acc);
    }
    int col = nt*16 + (lane & 15);
    int rb  = (lane >> 4) << 3;
    if (col < D_){
      float bkc = bk[col];
      #pragma unroll
      for (int r2 = 0; r2 < 8; ++r2){
        int gr = row0 + rb + r2;
        kx[(size_t)gr*KXS + col] = acc[r2]*w[gr] + bkc;
      }
    }
  }
}

// ---------------- score softmax + v_ts ----------------
__global__ void score_kernel(const float* kx, const float* w_att, const float* qdot,
                             float* vts){
  __shared__ float sc[L_];
  __shared__ float red[L_];
  int b = blockIdx.x, l = threadIdx.x;
  const float* kxb = kx + (size_t)b*L_*KXS;
  float dacc = 0.f;
  for (int d = 0; d < D_; ++d) dacc += kxb[(size_t)l*KXS + d]*w_att[d];
  float s = tanhf(dacc + qdot[b]);
  red[l] = s; __syncthreads();
  for (int o = 256; o; o >>= 1){ if (l < o) red[l] = fmaxf(red[l], red[l + o]); __syncthreads(); }
  float mx = red[0]; __syncthreads();
  float p = __expf(s - mx);
  red[l] = p; __syncthreads();
  for (int o = 256; o; o >>= 1){ if (l < o) red[l] += red[l + o]; __syncthreads(); }
  float sum = red[0]; __syncthreads();
  sc[l] = p / sum; __syncthreads();
  if (l < D_){
    float a = 0.f;
    for (int ll = 0; ll < L_; ++ll) a += sc[ll]*kxb[(size_t)ll*KXS + l];
    vts[(size_t)b*NK + l] = a;
  }
}

// ---------------- projection head ----------------
__global__ void head_kernel(const float* vts, const float* vs,
                            const float* Wproj, const float* bproj,
                            const float* Wm, const float* bm,
                            const float* Wd, const float* bd, float* out){
  __shared__ float tv[304], vns[304], vms[304], lg[4];
  int b = blockIdx.x, d = threadIdx.x;
  if (d < D_) tv[d] = vts[(size_t)b*NK + d];
  __syncthreads();
  if (d < D_){
    float a = bproj[d];
    for (int k = 0; k < D_; ++k) a += tv[k]*Wproj[(size_t)d*D_ + k];
    vns[d] = a + vs[(size_t)b*NK + d];
  }
  __syncthreads();
  if (d < D_){
    float a = bm[d];
    for (int k = 0; k < D_; ++k) a += vns[k]*Wm[(size_t)d*D_ + k];
    vms[d] = tanhf(a);
  }
  __syncthreads();
  if (d < P_){
    float a = bd[d];
    for (int k = 0; k < D_; ++k) a += vms[k]*Wd[(size_t)d*D_ + k];
    lg[d] = a;
  }
  __syncthreads();
  if (d == 0){
    float mx = fmaxf(lg[0], fmaxf(lg[1], lg[2]));
    float e0 = __expf(lg[0]-mx), e1 = __expf(lg[1]-mx), e2 = __expf(lg[2]-mx);
    float s = e0 + e1 + e2;
    out[b*P_ + 0] = e0/s; out[b*P_ + 1] = e1/s; out[b*P_ + 2] = e2/s;
  }
}

extern "C" void kernel_launch(void* const* d_in, const int* in_sizes, int n_in,
                              void* d_out, int out_size, void* d_ws, size_t ws_size,
                              hipStream_t stream) {
  const int*   text = (const int*)d_in[0];
  const int*   aidx = (const int*)d_in[1];
  const int*   xl   = (const int*)d_in[2];
  const int*   xr   = (const int*)d_in[3];
  const float* emb  = (const float*)d_in[4];
  const float* WihL = (const float*)d_in[5];
  const float* WhhL = (const float*)d_in[6];
  const float* bihL = (const float*)d_in[7];
  const float* bhhL = (const float*)d_in[8];
  const float* WihR = (const float*)d_in[9];
  const float* WhhR = (const float*)d_in[10];
  const float* bihR = (const float*)d_in[11];
  const float* bhhR = (const float*)d_in[12];
  const float* wl   = (const float*)d_in[13];
  const float* bl   = (const float*)d_in[14];
  const float* wr   = (const float*)d_in[15];
  const float* br   = (const float*)d_in[16];
  const float* Wk   = (const float*)d_in[17];
  const float* bk   = (const float*)d_in[18];
  const float* Wq   = (const float*)d_in[19];
  const float* bq   = (const float*)d_in[20];
  const float* watt = (const float*)d_in[21];
  const float* Wpj  = (const float*)d_in[22];
  const float* bpj  = (const float*)d_in[23];
  const float* Wm   = (const float*)d_in[24];
  const float* bm   = (const float*)d_in[25];
  const float* Wd   = (const float*)d_in[26];
  const float* bd   = (const float*)d_in[27];
  float* out = (float*)d_out;
  char* ws = (char*)d_ws;

  const size_t oLens  = 0;                       // 512 ints
  const size_t oPackG = 2048;                    // 4*NP*DK f16
  const size_t oPackK = oPackG + (size_t)4*NP*DK*2;   // +2,334,720
  const size_t oPbias = oPackK + (size_t)NK*DK*2;     // +194,560
  const size_t oAttn  = oPbias + (size_t)4*NP*4;      // +14,592
  const size_t oW     = oAttn + (size_t)2*B_*L_*4;    // +524,288
  const size_t oQdot  = oW + (size_t)B_*L_*4;         // +262,144
  const size_t oVs    = oQdot + 512;
  const size_t oVts   = oVs + (size_t)B_*NK*4;
  const size_t oKx    = oVts + (size_t)B_*NK*4;

  int*      lens  = (int*)(ws + oLens);
  _Float16* packG = (_Float16*)(ws + oPackG);
  _Float16* packK = (_Float16*)(ws + oPackK);
  float*    pbias = (float*)(ws + oPbias);
  float*    attn  = (float*)(ws + oAttn);
  float*    wbuf  = (float*)(ws + oW);
  float*    qdot  = (float*)(ws + oQdot);
  float*    vs    = (float*)(ws + oVs);
  float*    vts   = (float*)(ws + oVts);
  float*    kx    = (float*)(ws + oKx);

  lens_kernel<<<1, 512, 0, stream>>>(text, aidx, xl, xr, lens);
  pack_gru_kernel<<<(4*NP*DK + 255)/256, 256, 0, stream>>>(WihL, WhhL, WihR, WhhR, packG);
  pack_bias_kernel<<<(4*NP + 255)/256, 256, 0, stream>>>(bihL, bhhL, bihR, bhhR, pbias);
  pack_wk_kernel<<<(NK*DK + 255)/256, 256, 0, stream>>>(Wk, packK);
  aspect_kernel<<<B_, 512, 0, stream>>>(aidx, emb, lens, Wq, bq, watt, qdot);
  gru_kernel<<<16, 512, 0, stream>>>(xl, xr, emb, packG, pbias, lens, wl, bl, wr, br, attn);
  wpos_kernel<<<B_, L_, 0, stream>>>(lens, attn, wbuf);
  vs_kernel<<<B_, 320, 0, stream>>>(text, emb, wbuf, lens, vs);
  kx_kernel<<<(B_*L_)/16, 128, 0, stream>>>(text, emb, packK, bk, wbuf, kx);
  score_kernel<<<B_, L_, 0, stream>>>(kx, watt, qdot, vts);
  head_kernel<<<B_, 512, 0, stream>>>(vts, vs, Wpj, bpj, Wm, bm, Wd, bd, out);
}
